// EncodecQuantizer_67559835566227
// MI455X (gfx1250) — matrix operand order repaired and checked by reading, not compiled, not run on hardware
//
#include <hip/hip_runtime.h>

// ---- types for CDNA5 WMMA ----
typedef __attribute__((ext_vector_type(16))) __bf16 v16bf;
typedef __attribute__((ext_vector_type(8)))  float  v8f;

union FragA { v16bf v; unsigned short u[16]; };
union FragB { v16bf v; uint4 q[2]; };

// float -> bf16, round-to-nearest-even
__device__ inline unsigned short f2bf(float f) {
  unsigned int u = __float_as_uint(f);
  u += 0x7FFFu + ((u >> 16) & 1u);
  return (unsigned short)(u >> 16);
}

#define D      128
#define KCB    1024
#define NQ     8
#define BT     65536          // B*T = 16*4096
#define TOKW   128            // tokens per workgroup
#define RSTR   132            // padded f32 residual row stride (floats)
#define ESTR   136            // padded bf16 codebook-tile row stride (ushorts)
#define ROWS   32             // codes staged per round (two 16-wide N tiles)
#define NROUND (KCB / ROWS)   // 32 rounds per stage

// ---------------------------------------------------------------------------
// Prep: bf16 codebook copy + per-code squared norms. One wave32 per code.
// ---------------------------------------------------------------------------
__global__ void __launch_bounds__(256)
rvq_prep(const float* __restrict__ embed,
         unsigned short* __restrict__ ebf,
         float* __restrict__ norms) {
  const int gid  = blockIdx.x * 256 + threadIdx.x;
  const int code = gid >> 5;        // 0..8191
  const int lane = gid & 31;
  const float4 v = *(const float4*)(embed + (size_t)code * D + lane * 4);
  uint2 p;
  p.x = (unsigned)f2bf(v.x) | ((unsigned)f2bf(v.y) << 16);
  p.y = (unsigned)f2bf(v.z) | ((unsigned)f2bf(v.w) << 16);
  *(uint2*)(ebf + (size_t)code * D + lane * 4) = p;
  float s = v.x * v.x + v.y * v.y + v.z * v.z + v.w * v.w;
  #pragma unroll
  for (int mm = 16; mm >= 1; mm >>= 1) s += __shfl_xor(s, mm, 32);
  if (lane == 0) norms[code] = s;
}

// ---------------------------------------------------------------------------
// Main RVQ kernel: 512 WGs x 256 threads (8 wave32). Each WG owns 128 tokens.
// Double-buffered B tiles, register-staged prefetch (tiles + norms),
// 16 ds_loads up front then an 8-WMMA burst per round.
// ---------------------------------------------------------------------------
__global__ void __launch_bounds__(256)
rvq_main(const float* __restrict__ x,
         const float* __restrict__ embed,
         const unsigned short* __restrict__ ebf,
         const float* __restrict__ norms,
         int* __restrict__ codes) {
  extern __shared__ char smem[];
  float*          sResid = (float*)smem;                              // TOKW*RSTR f32
  unsigned short* sEb0   = (unsigned short*)(smem + TOKW * RSTR * 4); // ROWS*ESTR bf16
  unsigned short* sEb1   = sEb0 + ROWS * ESTR;                        // ROWS*ESTR bf16
  int*            sIdx   = (int*)(smem + TOKW * RSTR * 4 + 2 * ROWS * ESTR * 2);

  const int tid     = threadIdx.x;
  const int lane    = tid & 31;
  const int w       = tid >> 5;     // wave id 0..7
  const int m       = lane & 15;    // row/col within 16
  const int g       = lane >> 4;    // half-wave
  const int tokBase = blockIdx.x * TOKW;

  // per-thread staging slots: 1024 uint2 per 32x128 bf16 tile, 4 per thread
  int srow[4], sco[4];
  #pragma unroll
  for (int k = 0; k < 4; ++k) {
    int i = tid + k * 256;
    srow[k] = i >> 5;          // 0..31 (code row within tile)
    sco[k]  = (i & 31) * 4;    // bf16 column 0..124
  }

  // load residual tile (f32) into LDS
  for (int i = tid; i < TOKW * (D / 4); i += 256) {
    int token = i >> 5, d4 = i & 31;
    float4 v = ((const float4*)x)[(tokBase + token) * (D / 4) + d4];
    *(float4*)(sResid + token * RSTR + d4 * 4) = v;
  }
  __syncthreads();

  const float* myRow = sResid + (w * 16 + m) * RSTR;

  for (int q = 0; q < NQ; ++q) {
    // ---- bf16 A fragments (ISA 16-bit A 16x32 layout), rebuilt per stage ----
    FragA a[4];
    #pragma unroll
    for (int c = 0; c < 4; ++c) {
      const float* p = myRow + c * 32 + 8 * g;
      float4 f0 = *(const float4*)(p);
      float4 f1 = *(const float4*)(p + 4);
      float4 f2 = *(const float4*)(p + 16);
      float4 f3 = *(const float4*)(p + 20);
      a[c].u[0]  = f2bf(f0.x); a[c].u[1]  = f2bf(f0.y);
      a[c].u[2]  = f2bf(f0.z); a[c].u[3]  = f2bf(f0.w);
      a[c].u[4]  = f2bf(f1.x); a[c].u[5]  = f2bf(f1.y);
      a[c].u[6]  = f2bf(f1.z); a[c].u[7]  = f2bf(f1.w);
      a[c].u[8]  = f2bf(f2.x); a[c].u[9]  = f2bf(f2.y);
      a[c].u[10] = f2bf(f2.z); a[c].u[11] = f2bf(f2.w);
      a[c].u[12] = f2bf(f3.x); a[c].u[13] = f2bf(f3.y);
      a[c].u[14] = f2bf(f3.z); a[c].u[15] = f2bf(f3.w);
    }

    float bestv[8];
    int   besti[8];
    #pragma unroll
    for (int r = 0; r < 8; ++r) { bestv[r] = -3.402823466e38f; besti[r] = 0; }

    const unsigned short* ebq = ebf + (size_t)q * KCB * D;
    const float*          nq  = norms + q * KCB;

    // preload round 0 tile into sEb0 and round 0 norms into registers
    #pragma unroll
    for (int k = 0; k < 4; ++k) {
      uint2 d = *(const uint2*)(ebq + srow[k] * D + sco[k]);
      *(uint2*)(sEb0 + srow[k] * ESTR + sco[k]) = d;
    }
    float nrm0 = nq[m];
    float nrm1 = nq[16 + m];
    __syncthreads();

    for (int t = 0; t < NROUND; ++t) {
      const unsigned short* cur = (t & 1) ? sEb1 : sEb0;
      unsigned short*       nxt = (t & 1) ? sEb0 : sEb1;
      const int tn = (t + 1) & (NROUND - 1);   // wraps on last round (harmless)

      // ---- issue next-round global prefetch first (covered by WMMA burst) ----
      uint2 pre[4];
      #pragma unroll
      for (int k = 0; k < 4; ++k)
        pre[k] = *(const uint2*)(ebq + (tn * ROWS + srow[k]) * D + sco[k]);
      float nrm0n = nq[tn * ROWS + m];
      float nrm1n = nq[tn * ROWS + 16 + m];

      // ---- load ALL 8 B fragments, then one 8-WMMA burst ----
      FragB b[8];
      #pragma unroll
      for (int c = 0; c < 4; ++c) {
        const unsigned short* bp0 = cur + m * ESTR        + c * 32 + 8 * g;
        const unsigned short* bp1 = cur + (m + 16) * ESTR + c * 32 + 8 * g;
        b[2 * c].q[0]     = *(const uint4*)(bp0);
        b[2 * c].q[1]     = *(const uint4*)(bp0 + 16);
        b[2 * c + 1].q[0] = *(const uint4*)(bp1);
        b[2 * c + 1].q[1] = *(const uint4*)(bp1 + 16);
      }

      v8f acc0 = {0.f, 0.f, 0.f, 0.f, 0.f, 0.f, 0.f, 0.f};
      v8f acc1 = {0.f, 0.f, 0.f, 0.f, 0.f, 0.f, 0.f, 0.f};
      #pragma unroll
      for (int c = 0; c < 4; ++c) {
        acc0 = __builtin_amdgcn_wmma_f32_16x16x32_bf16(
            false, a[c].v, false, b[2 * c].v,     (short)0, acc0, false, false);
        acc1 = __builtin_amdgcn_wmma_f32_16x16x32_bf16(
            false, a[c].v, false, b[2 * c + 1].v, (short)0, acc1, false, false);
      }

      // drain prefetched tile into the other LDS buffer
      #pragma unroll
      for (int k = 0; k < 4; ++k)
        *(uint2*)(nxt + srow[k] * ESTR + sco[k]) = pre[k];

      // running argmax of 2 r.e - ||e||^2 (first-index-wins on ties);
      // norms were prefetched last round -> no load wait here
      const int c0 = t * ROWS + m;
      const int c1 = c0 + 16;
      #pragma unroll
      for (int r = 0; r < 8; ++r) {
        float v0 = 2.0f * acc0[r] - nrm0;
        if (v0 > bestv[r]) { bestv[r] = v0; besti[r] = c0; }
      }
      #pragma unroll
      for (int r = 0; r < 8; ++r) {
        float v1 = 2.0f * acc1[r] - nrm1;
        if (v1 > bestv[r]) { bestv[r] = v1; besti[r] = c1; }
      }
      nrm0 = nrm0n;
      nrm1 = nrm1n;
      __syncthreads();   // single barrier per round
    }

    // reduce across the 16 code-columns (lanes) of each half-wave
    #pragma unroll
    for (int s = 1; s < 16; s <<= 1) {
      #pragma unroll
      for (int r = 0; r < 8; ++r) {
        float ov = __shfl_xor(bestv[r], s, 32);
        int   oi = __shfl_xor(besti[r], s, 32);
        if (ov > bestv[r] || (ov == bestv[r] && oi < besti[r])) {
          bestv[r] = ov; besti[r] = oi;
        }
      }
    }

    // token(lane,r) = w*16 + g*8 + r (C/D VGPR layout); lane m==0 of each half writes
    if (m == 0) {
      #pragma unroll
      for (int r = 0; r < 8; ++r) {
        int tl = w * 16 + g * 8 + r;
        sIdx[tl] = besti[r];
        codes[q * BT + tokBase + tl] = besti[r];  // [N_Q, B, T] layout
      }
    }
    __syncthreads();

    // residual -= embed[q][code]  (f32 precision)
    const float* eq = embed + (size_t)q * KCB * D;
    for (int i = tid; i < TOKW * (D / 4); i += 256) {
      int token = i >> 5, d4 = i & 31;
      int ci = sIdx[token];
      float4 ev = *(const float4*)(eq + ci * D + d4 * 4);
      float4* p = (float4*)(sResid + token * RSTR + d4 * 4);
      float4 rv = *p;
      rv.x -= ev.x; rv.y -= ev.y; rv.z -= ev.z; rv.w -= ev.w;
      *p = rv;
    }
    __syncthreads();
  }
}

extern "C" void kernel_launch(void* const* d_in, const int* in_sizes, int n_in,
                              void* d_out, int out_size, void* d_ws, size_t ws_size,
                              hipStream_t stream) {
  (void)in_sizes; (void)n_in; (void)out_size; (void)ws_size;
  const float* x     = (const float*)d_in[0];   // [16,4096,128] f32
  const float* embed = (const float*)d_in[1];   // [8,1024,128] f32

  unsigned short* ebf   = (unsigned short*)d_ws;                            // 2 MB bf16 codebook
  float*          norms = (float*)((char*)d_ws + (size_t)NQ * KCB * D * 2); // 32 KB norms
  int*            codes = (int*)d_out;                                      // [8,16,4096] int32

  rvq_prep<<<1024, 256, 0, stream>>>(embed, ebf, norms);

  size_t shmem = (size_t)TOKW * RSTR * 4 + 2 * ROWS * ESTR * 2 + TOKW * 4;  // 85504 B
  rvq_main<<<BT / TOKW, 256, shmem, stream>>>(x, embed, ebf, norms, codes);
}
